// MultiModalGCN_76218489635510
// MI455X (gfx1250) — matrix operand order repaired and checked by reading, not compiled, not run on hardware
//
#include <hip/hip_runtime.h>
#include <hip/hip_bf16.h>

// ---------------------------------------------------------------------------
// MultiModalGCN on MI455X (gfx1250, wave32, WMMA, LDS-staged B, f32 atomics)
//   m  = relu(mel @ W_mel + b_mel)                  -- WMMA bf16, K=16384
//   x  = relu([text|m] @ W_cat + b_cat)             -- WMMA bf16, K=896 (split)
//   h1 = x @ W1 ; y1 = b1 + sym-norm aggregate(h1)  -- WMMA + f32 atomics
//   h2 = relu(y1) @ W2 ; out = b2 + aggregate(h2)   -- WMMA + f32 atomics
// ---------------------------------------------------------------------------

typedef __attribute__((ext_vector_type(16))) __bf16 v16bf;
typedef __attribute__((ext_vector_type(8)))  float  v8f;

#define HID      128
#define TEXT_DIM 768
#define MEL_K    16384
#define NCLS     4

__device__ __forceinline__ float4 relu4(float4 v) {
  v.x = fmaxf(v.x, 0.f); v.y = fmaxf(v.y, 0.f);
  v.z = fmaxf(v.z, 0.f); v.w = fmaxf(v.w, 0.f);
  return v;
}

// A-fragment (16x32 bf16, ISA 7.12.2 layout): lane L (row = L&15) holds
//   elements 0..7  = K(kb+0 .. kb+7),   kb = (L<16 ? 0 : 8)
//   elements 8..15 = K(kb+16 .. kb+23)
__device__ __forceinline__ v16bf load_a_frag(const float* __restrict__ A, int lda,
                                             int row, int k0, int lane, bool relu_in) {
  const float* p = A + (size_t)row * lda + k0 + ((lane >> 4) << 3);
  float4 f0 = *(const float4*)(p + 0);
  float4 f1 = *(const float4*)(p + 4);
  float4 f2 = *(const float4*)(p + 16);
  float4 f3 = *(const float4*)(p + 20);
  if (relu_in) { f0 = relu4(f0); f1 = relu4(f1); f2 = relu4(f2); f3 = relu4(f3); }
  v16bf a;
  a[0]  = (__bf16)f0.x; a[1]  = (__bf16)f0.y; a[2]  = (__bf16)f0.z; a[3]  = (__bf16)f0.w;
  a[4]  = (__bf16)f1.x; a[5]  = (__bf16)f1.y; a[6]  = (__bf16)f1.z; a[7]  = (__bf16)f1.w;
  a[8]  = (__bf16)f2.x; a[9]  = (__bf16)f2.y; a[10] = (__bf16)f2.z; a[11] = (__bf16)f2.w;
  a[12] = (__bf16)f3.x; a[13] = (__bf16)f3.y; a[14] = (__bf16)f3.z; a[15] = (__bf16)f3.w;
  return a;
}

// ---------------------------------------------------------------------------
// GEMM: out[M,NSTORE] = act( A[M,KTOT](f32) x B[KTOT, NTILES*16] + bias )
// BT is bf16, transposed [NTILES*16, KTOT].
// B is staged per-block in LDS in *fragment order*: chunk c (= tile t*32+lane L)
// holds the 16 contiguous bf16 (32B) that lane L of tile t consumes:
//   global row n = t*16 + (L&15), i.e. BT[n*KTOT + k0 + (L>>4)*8 + {0..7, 16..23}].
// Double-buffered, one barrier per K-step; next chunk's global loads issue
// before the WMMAs on the current chunk so L2 latency hides under compute.
// All B fragments are pulled into distinct registers first so the ds_loads
// clause together and the 8 WMMAs issue back-to-back after one s_wait_dscnt.
// ---------------------------------------------------------------------------
template<int KTOT, int NTILES, int NSTORE, bool RELU_OUT, bool RELU_IN>
__global__ __launch_bounds__(256) void gemm_bf16_wmma(
    const float* __restrict__ A, int lda,
    const __bf16* __restrict__ BT,
    const float* __restrict__ bias,
    float* __restrict__ out, int M)
{
  constexpr int CHUNKS = NTILES * 32;          // 32B chunks per K-step
  constexpr int ITERS  = KTOT / 32;
  __shared__ __bf16 smem[2][CHUNKS * 16];

  const int tid  = threadIdx.x;
  const int lane = tid & 31;
  const int wave = blockIdx.x * (blockDim.x >> 5) + (tid >> 5);
  const bool active = (wave << 4) < M;
  const int row0 = active ? (wave << 4) : 0;   // clamp: all waves run the loop

  // B-fill role for this thread
  const bool filler = tid < CHUNKS;
  const int  bt = tid >> 5, bl = tid & 31;
  const __bf16* bsrc = BT + (size_t)(bt * 16 + (bl & 15)) * KTOT + ((bl >> 4) << 3);

  uint4 r0 = {}, r1 = {};
  if (filler) { r0 = *(const uint4*)(bsrc); r1 = *(const uint4*)(bsrc + 16); }
  if (filler) {
    *(uint4*)(&smem[0][tid * 16])     = r0;
    *(uint4*)(&smem[0][tid * 16 + 8]) = r1;
  }
  __syncthreads();

  v8f acc[NTILES] = {};
  const int arow = row0 + (lane & 15);

  for (int it = 0; it < ITERS; ++it) {
    const int k0 = it * 32;
    // issue next B chunk's global loads early (latency hidden under WMMAs)
    if (filler && it + 1 < ITERS) {
      r0 = *(const uint4*)(bsrc + (it + 1) * 32);
      r1 = *(const uint4*)(bsrc + (it + 1) * 32 + 16);
    }
    if (KTOT > 4096 && k0 + 128 < KTOT)        // prefetch A stream (mel GEMM)
      __builtin_prefetch(A + (size_t)arow * lda + k0 + 128, 0, 3);

    const v16bf a = load_a_frag(A, lda, arow, k0, lane, RELU_IN);
    const __bf16* sb = &smem[it & 1][lane * 16];
    v16bf bfrag[NTILES];
#pragma unroll
    for (int t = 0; t < NTILES; ++t)
      bfrag[t] = *(const v16bf*)(sb + t * 512);      // (t*32+lane)*16
#pragma unroll
    for (int t = 0; t < NTILES; ++t)
      acc[t] = __builtin_amdgcn_wmma_f32_16x16x32_bf16(
          false, a, false, bfrag[t], (short)0, acc[t], false, false);

    if (filler && it + 1 < ITERS) {
      *(uint4*)(&smem[(it + 1) & 1][tid * 16])     = r0;
      *(uint4*)(&smem[(it + 1) & 1][tid * 16 + 8]) = r1;
    }
    __syncthreads();
  }

  if (active) {
    const int bcol  = lane & 15;
    const int rbase = row0 + ((lane >> 4) << 3);  // D: VGPR i -> row i + 8*(L>=16)
#pragma unroll
    for (int t = 0; t < NTILES; ++t) {
      const int nc = t * 16 + bcol;
      if (nc < NSTORE) {
        const float bv = bias ? bias[nc] : 0.f;
#pragma unroll
        for (int i = 0; i < 8; ++i) {
          float v = acc[t][i] + bv;
          if (RELU_OUT) v = fmaxf(v, 0.f);
          out[(size_t)(rbase + i) * NSTORE + nc] = v;
        }
      }
    }
  }
}

// Fused concat GEMM: x = relu([text | m] @ W_cat + b_cat), K = 768 + 128
__global__ __launch_bounds__(256) void gemm_cat_wmma(
    const float* __restrict__ text, const float* __restrict__ m,
    const __bf16* __restrict__ BT /* [128, 896] bf16 */,
    const float* __restrict__ bias, float* __restrict__ out, int M)
{
  constexpr int KTOT   = TEXT_DIM + HID;
  constexpr int CHUNKS = 8 * 32;
  constexpr int ITERS  = KTOT / 32;
  __shared__ __bf16 smem[2][CHUNKS * 16];

  const int tid  = threadIdx.x;
  const int lane = tid & 31;
  const int wave = blockIdx.x * (blockDim.x >> 5) + (tid >> 5);
  const bool active = (wave << 4) < M;
  const int row0 = active ? (wave << 4) : 0;

  const bool filler = tid < CHUNKS;
  const int  bt = tid >> 5, bl = tid & 31;
  const __bf16* bsrc = BT + (size_t)(bt * 16 + (bl & 15)) * KTOT + ((bl >> 4) << 3);

  uint4 r0 = {}, r1 = {};
  if (filler) { r0 = *(const uint4*)(bsrc); r1 = *(const uint4*)(bsrc + 16); }
  if (filler) {
    *(uint4*)(&smem[0][tid * 16])     = r0;
    *(uint4*)(&smem[0][tid * 16 + 8]) = r1;
  }
  __syncthreads();

  v8f acc[8] = {};
  const int arow = row0 + (lane & 15);

  for (int it = 0; it < ITERS; ++it) {
    const int k0 = it * 32;
    if (filler && it + 1 < ITERS) {
      r0 = *(const uint4*)(bsrc + (it + 1) * 32);
      r1 = *(const uint4*)(bsrc + (it + 1) * 32 + 16);
    }
    const v16bf a = (k0 < TEXT_DIM)
        ? load_a_frag(text, TEXT_DIM, arow, k0, lane, false)
        : load_a_frag(m, HID, arow, k0 - TEXT_DIM, lane, false);
    const __bf16* sb = &smem[it & 1][lane * 16];
    v16bf bfrag[8];
#pragma unroll
    for (int t = 0; t < 8; ++t)
      bfrag[t] = *(const v16bf*)(sb + t * 512);
#pragma unroll
    for (int t = 0; t < 8; ++t)
      acc[t] = __builtin_amdgcn_wmma_f32_16x16x32_bf16(
          false, a, false, bfrag[t], (short)0, acc[t], false, false);

    if (filler && it + 1 < ITERS) {
      *(uint4*)(&smem[(it + 1) & 1][tid * 16])     = r0;
      *(uint4*)(&smem[(it + 1) & 1][tid * 16 + 8]) = r1;
    }
    __syncthreads();
  }

  if (active) {
    const int bcol  = lane & 15;
    const int rbase = row0 + ((lane >> 4) << 3);
#pragma unroll
    for (int t = 0; t < 8; ++t) {
      const int nc = t * 16 + bcol;
      const float bv = bias[nc];
#pragma unroll
      for (int i = 0; i < 8; ++i)
        out[(size_t)(rbase + i) * HID + nc] = fmaxf(acc[t][i] + bv, 0.f);
    }
  }
}

// ---- weight prep: W[K,Nin] f32 -> WT[Nout,K] bf16 (rows >= Nin zero-padded)
__global__ void transpose_bf16(const float* __restrict__ W, __bf16* __restrict__ out,
                               int K, int Nin, int Nout) {
  int idx = blockIdx.x * blockDim.x + threadIdx.x;
  if (idx >= K * Nout) return;
  int n = idx / K, k = idx - n * K;
  out[idx] = (n < Nin) ? (__bf16)W[(size_t)k * Nin + n] : (__bf16)0.0f;
}

// ---- degree / normalization
__global__ void deg_init(float* deg, int n) {
  int i = blockIdx.x * blockDim.x + threadIdx.x;
  if (i < n) deg[i] = 1.0f;                         // self-loop
}
__global__ void deg_count(const int* __restrict__ dst, float* deg, int e) {
  int i = blockIdx.x * blockDim.x + threadIdx.x;
  if (i < e) atomicAdd(&deg[dst[i]], 1.0f);
}
__global__ void deg_finish(float* deg, int n) {
  int i = blockIdx.x * blockDim.x + threadIdx.x;
  if (i < n) deg[i] = rsqrtf(deg[i]);               // deg >= 1 always
}

// ---- y[i,c] = bias[c] + h[i,c] * dinv[i]^2   (self-loop term + bias init)
__global__ void self_init(const float* __restrict__ h, const float* __restrict__ dinv,
                          const float* __restrict__ bias, float* __restrict__ y,
                          int n, int F) {
  int idx = blockIdx.x * blockDim.x + threadIdx.x;
  if (idx >= n * F) return;
  int i = idx / F, c = idx - i * F;
  float di = dinv[i];
  y[idx] = bias[c] + h[idx] * di * di;
}

// ---- edge scatter: y[d] += h[s] * dinv[s]*dinv[d], float4 per thread
__global__ void scatter_edges(const int* __restrict__ src, const int* __restrict__ dst,
                              const float* __restrict__ dinv, const float* __restrict__ h,
                              float* __restrict__ y, int E, int F) {
  int idx = blockIdx.x * blockDim.x + threadIdx.x;
  const int groups = F >> 2;
  int e = idx / groups, g = idx - e * groups;
  if (e >= E) return;
  int s = src[e], d = dst[e];
  float norm = dinv[s] * dinv[d];
  float4 hv = *(const float4*)(h + (size_t)s * F + (g << 2));
  float* yp = y + (size_t)d * F + (g << 2);
  atomicAdd(yp + 0, hv.x * norm);
  atomicAdd(yp + 1, hv.y * norm);
  atomicAdd(yp + 2, hv.z * norm);
  atomicAdd(yp + 3, hv.w * norm);
}

extern "C" void kernel_launch(void* const* d_in, const int* in_sizes, int n_in,
                              void* d_out, int out_size, void* d_ws, size_t ws_size,
                              hipStream_t stream) {
  const float* text  = (const float*)d_in[0];
  const float* mel   = (const float*)d_in[1];
  const int*   ei    = (const int*)  d_in[2];
  const float* W_mel = (const float*)d_in[3];
  const float* b_mel = (const float*)d_in[4];
  const float* W_cat = (const float*)d_in[5];
  const float* b_cat = (const float*)d_in[6];
  const float* W1    = (const float*)d_in[7];
  const float* b1    = (const float*)d_in[8];
  const float* W2    = (const float*)d_in[9];
  const float* b2    = (const float*)d_in[10];

  const int n = in_sizes[0] / TEXT_DIM;     // 20000
  const int E = in_sizes[2] / 2;            // 640000
  const int* src = ei;
  const int* dst = ei + E;

  // workspace carve-up (256B aligned slots)
  char* ws = (char*)d_ws;
  size_t off = 0;
  auto carve = [&](size_t bytes) -> void* {
    void* p = ws + off;
    off = (off + bytes + 255) & ~(size_t)255;
    return p;
  };
  float*  dinv  = (float*) carve((size_t)n * 4);
  float*  mfeat = (float*) carve((size_t)n * HID * 4);
  float*  x     = (float*) carve((size_t)n * HID * 4);
  float*  h1    = (float*) carve((size_t)n * HID * 4);
  float*  y1    = (float*) carve((size_t)n * HID * 4);
  float*  h2    = (float*) carve((size_t)n * NCLS * 4);
  __bf16* WmelT = (__bf16*)carve((size_t)HID * MEL_K * 2);
  __bf16* WcatT = (__bf16*)carve((size_t)HID * (TEXT_DIM + HID) * 2);
  __bf16* W1T   = (__bf16*)carve((size_t)HID * HID * 2);
  __bf16* W2T   = (__bf16*)carve((size_t)16 * HID * 2);
  float*  outF  = (float*)d_out;

  const int B = 256;
  auto blocks = [](long total, int b) { return (int)((total + b - 1) / b); };

  // weight prep (bf16, transposed, W2 padded to 16 output cols)
  transpose_bf16<<<blocks((long)MEL_K * HID, B), B, 0, stream>>>(W_mel, WmelT, MEL_K, HID, HID);
  transpose_bf16<<<blocks((long)(TEXT_DIM + HID) * HID, B), B, 0, stream>>>(W_cat, WcatT, TEXT_DIM + HID, HID, HID);
  transpose_bf16<<<blocks((long)HID * HID, B), B, 0, stream>>>(W1, W1T, HID, HID, HID);
  transpose_bf16<<<blocks((long)HID * 16, B), B, 0, stream>>>(W2, W2T, HID, NCLS, 16);

  // symmetric normalization: dinv = rsqrt(1 + indegree)
  deg_init<<<blocks(n, B), B, 0, stream>>>(dinv, n);
  deg_count<<<blocks(E, B), B, 0, stream>>>(dst, dinv, E);
  deg_finish<<<blocks(n, B), B, 0, stream>>>(dinv, n);

  const int gemm_grid = blocks((long)(n / 16), 8);   // 8 waves/block, 16 rows/wave

  // m = relu(mel @ W_mel + b_mel)
  gemm_bf16_wmma<MEL_K, 8, HID, true, false>
      <<<gemm_grid, B, 0, stream>>>(mel, MEL_K, WmelT, b_mel, mfeat, n);
  // x = relu([text|m] @ W_cat + b_cat)
  gemm_cat_wmma<<<gemm_grid, B, 0, stream>>>(text, mfeat, WcatT, b_cat, x, n);

  // GCN layer 1
  gemm_bf16_wmma<HID, 8, HID, false, false>
      <<<gemm_grid, B, 0, stream>>>(x, HID, W1T, nullptr, h1, n);
  self_init<<<blocks((long)n * HID, B), B, 0, stream>>>(h1, dinv, b1, y1, n, HID);
  scatter_edges<<<blocks((long)E * (HID / 4), B), B, 0, stream>>>(src, dst, dinv, h1, y1, E, HID);

  // GCN layer 2 (relu applied to y1 on A-load; W2 padded to 16 cols, store 4)
  gemm_bf16_wmma<HID, 1, NCLS, false, true>
      <<<gemm_grid, B, 0, stream>>>(y1, HID, W2T, nullptr, h2, n);
  self_init<<<blocks((long)n * NCLS, B), B, 0, stream>>>(h2, dinv, b2, outF, n, NCLS);
  scatter_edges<<<blocks((long)E, B), B, 0, stream>>>(src, dst, dinv, h2, outF, E, NCLS);
}
